// MaskedLinearMultivariate_62852551409794
// MI455X (gfx1250) — matrix operand order repaired and checked by reading, not compiled, not run on hardware
//
#include <hip/hip_runtime.h>
#include <stdint.h>

typedef __attribute__((ext_vector_type(16))) __bf16 v16bf;
typedef __attribute__((ext_vector_type(8)))  float  v8f;

#define Bsz 64
#define Gsz 8
#define Nsz 88
#define Tsz 64
#define Dsz 88
#define Osz 64
#define Ksz (Nsz * Tsz)     /* 5632 : K-length per g          */
#define COLS (Dsz * Osz)    /* 5632 : output columns (d,o)    */
#define PITCH 144           /* LDS row pitch: 64 bf16 = 128B data + 16B pad */

// round-half-up f32->bf16, packed pair in one v_perm_b32
__device__ __forceinline__ uint32_t pack2_bf16(float lo, float hi) {
    uint32_t ul = __float_as_uint(lo) + 0x8000u;
    uint32_t uh = __float_as_uint(hi) + 0x8000u;
    return __builtin_amdgcn_perm(uh, ul, 0x07060302u);   // {uh[31:16], ul[31:16]}
}

// ---------------------------------------------------------------------------
// Kernel 1: A[g][b][k] = bf16( x[b,n,t] * y[b,g] ),  k = n*64+t, packed pairs.
// ---------------------------------------------------------------------------
__global__ void prep_A_kernel(const float* __restrict__ x,
                              const float* __restrict__ y,
                              uint32_t* __restrict__ A) {
    int p = blockIdx.x * blockDim.x + threadIdx.x;   // pair index
    int e = p * 2;
    if (e >= Gsz * Bsz * Ksz) return;
    int g   = e / (Bsz * Ksz);
    int rem = e - g * (Bsz * Ksz);
    int b   = rem / Ksz;
    int k   = rem - b * Ksz;
    float yv = y[b * Gsz + g];
    float x0 = x[b * Ksz + k];
    float x1 = x[b * Ksz + k + 1];
    A[p] = pack2_bf16(x0 * yv, x1 * yv);
}

// ---------------------------------------------------------------------------
// Kernel 2: per-(d,g) GEMM tile: partial[g][b][d*64+o] = sum_k A_g[b,k]*Wm[k,o]
// Block = 512 threads = 16 wave32; each wave owns one 16x16 C tile.
// K-step 64, double-buffered LDS, one barrier per step, 2 WMMA/wave/step.
// ---------------------------------------------------------------------------
__global__ __launch_bounds__(512) void gemm_bf16_kernel(
    const float* __restrict__ W,        // (G,D,O,N,T) f32
    const int*   __restrict__ mask,     // (O,T) int
    const uint32_t* __restrict__ A,     // bf16 pairs, [g][b][k]
    float* __restrict__ partial)        // [g][b][COLS]
{
    __shared__ __align__(16) unsigned char AB[2 * 64 * PITCH];  // 18 KB (dbl buf)
    __shared__ __align__(16) unsigned char BB[2 * 64 * PITCH];  // 18 KB (dbl buf)
    const int BUFB = 64 * PITCH;

    const int tid = threadIdx.x;
    const int d   = blockIdx.x;         // 0..87
    const int g   = blockIdx.y;         // 0..7

    const int wave  = tid >> 5;
    const int lane  = tid & 31;
    const int rtile = wave >> 2;        // 0..3 -> b rows 16*rtile..
    const int ctile = wave & 3;         // 0..3 -> o cols 16*ctile..

    const float* wbase = W + ((size_t)(g * Dsz + d) * Osz) * Ksz;
    const uint4* abase = (const uint4*)(A + ((size_t)g * Bsz * Ksz) / 2);
    const int    AROWQ = Ksz / 8;       // 704 uint4 per A row

    // cooperative loader mapping: 8 threads per column/row,
    // each thread owns 8 CONSECUTIVE k (32B global, one uint4 to LDS)
    const int bcol = tid >> 3;          // 0..63 : o column
    const int bseg = tid & 7;           // k-offset = bseg*8 .. +7
    const int arow = bcol;              // 0..63 : b row
    const int aseg = bseg;              // one uint4 = 16B per thread

    // loop-invariant mask multipliers: t == bseg*8 + j for every k-step
    float m[8];
    {
        const int* mp = mask + bcol * Tsz + bseg * 8;
        #pragma unroll
        for (int j = 0; j < 8; ++j) m[j] = (float)mp[j];
    }

    // per-lane LDS fragment read pointers (ISA bf16 WMMA layouts)
    const int m16 = lane & 15;
    const int kg  = lane >> 4;
    const unsigned char* aRd = AB + (rtile * 16 + m16) * PITCH + kg * 16; // +{0,32}
    const unsigned char* bRd = BB + (ctile * 16 + m16) * PITCH + kg * 16; // +{0,32}

    v8f acc = {};

    // software pipeline: prologue global loads for k0 = 0
    float4 wv0 = *(const float4*)(wbase + (size_t)bcol * Ksz + bseg * 8);
    float4 wv1 = *(const float4*)(wbase + (size_t)bcol * Ksz + bseg * 8 + 4);
    uint4  av  = *(abase + (size_t)arow * AROWQ + aseg);

    for (int k0 = 0; k0 < Ksz; k0 += 64) {
        const int bo = ((k0 >> 6) & 1) * BUFB;

        // mask + convert current 64-k chunk, stage to LDS (one b128 each)
        uint4 pw;
        pw.x = pack2_bf16(wv0.x * m[0], wv0.y * m[1]);
        pw.y = pack2_bf16(wv0.z * m[2], wv0.w * m[3]);
        pw.z = pack2_bf16(wv1.x * m[4], wv1.y * m[5]);
        pw.w = pack2_bf16(wv1.z * m[6], wv1.w * m[7]);
        *(uint4*)(BB + bo + bcol * PITCH + bseg * 16) = pw;
        *(uint4*)(AB + bo + arow * PITCH + aseg * 16) = av;
        __syncthreads();

        // issue next chunk's global loads (overlap with WMMA below)
        if (k0 + 64 < Ksz) {
            const size_t wk = (size_t)bcol * Ksz + (k0 + 64) + bseg * 8;
            wv0 = *(const float4*)(wbase + wk);
            wv1 = *(const float4*)(wbase + wk + 4);
            av  = *(abase + (size_t)arow * AROWQ + (k0 + 64) / 8 + aseg);
            if (k0 + 320 < Ksz)   // prefetch the weight stream 4 steps ahead
                __builtin_prefetch(wbase + (size_t)bcol * Ksz + (k0 + 256) + bseg * 8, 0, 0);
        }

        // two 16x16x32 WMMAs covering K = [k0, k0+64)
        #pragma unroll
        for (int s = 0; s < 2; ++s) {
            union Frag { uint4 u[2]; v16bf v; } afr, bfr;
            const unsigned char* ap = aRd + bo + s * 64;
            const unsigned char* bp = bRd + bo + s * 64;
            afr.u[0] = *(const uint4*)(ap);
            afr.u[1] = *(const uint4*)(ap + 32);
            bfr.u[0] = *(const uint4*)(bp);
            bfr.u[1] = *(const uint4*)(bp + 32);
            acc = __builtin_amdgcn_wmma_f32_16x16x32_bf16(
                      false, afr.v, false, bfr.v, (short)0, acc, false, false);
        }
    }

    // scatter C tile: lane L -> col = L&15, rows m = (L>>4)*8 + i
    const int ncol = lane & 15;
    const int mb   = (lane >> 4) * 8;
    float* pb = partial + (size_t)g * Bsz * COLS
                        + (size_t)(rtile * 16 + mb) * COLS
                        + d * Osz + ctile * 16 + ncol;
    #pragma unroll
    for (int i = 0; i < 8; ++i) pb[(size_t)i * COLS] = acc[i];
}

// ---------------------------------------------------------------------------
// Kernel 3: out[b,d,o] = bias[d,o] + sum_g partial[g][b][col]
// ---------------------------------------------------------------------------
__global__ void reduce_bias_kernel(const float* __restrict__ partial,
                                   const float* __restrict__ bias,
                                   float* __restrict__ out) {
    int i = blockIdx.x * blockDim.x + threadIdx.x;
    if (i >= Bsz * COLS) return;
    int col = i % COLS;
    float s = bias[col];
    #pragma unroll
    for (int g = 0; g < Gsz; ++g)
        s += partial[(size_t)g * Bsz * COLS + i];
    out[i] = s;
}

// ---------------------------------------------------------------------------
extern "C" void kernel_launch(void* const* d_in, const int* in_sizes, int n_in,
                              void* d_out, int out_size, void* d_ws, size_t ws_size,
                              hipStream_t stream) {
    const float* x      = (const float*)d_in[0];   // (B,N,T)
    const float* y      = (const float*)d_in[1];   // (B,G)
    const float* weight = (const float*)d_in[2];   // (G,D,O,N,T)
    const float* bias   = (const float*)d_in[3];   // (D,O)
    const int*   mask   = (const int*)  d_in[4];   // (O,T)
    float* out = (float*)d_out;

    const size_t A_BYTES = (size_t)Gsz * Bsz * Ksz * 2;          // 5,767,168
    uint32_t* A_bf = (uint32_t*)d_ws;
    float*    part = (float*)((char*)d_ws + A_BYTES);            // 11,534,336 B

    // 1) A = bf16(x * y), packed
    {
        int pairs = Gsz * Bsz * Ksz / 2;                         // 1,441,792
        prep_A_kernel<<<(pairs + 255) / 256, 256, 0, stream>>>(x, y, A_bf);
    }
    // 2) 704 tile-GEMMs streaming the 1 GB weight once
    {
        dim3 grid(Dsz, Gsz);
        gemm_bf16_kernel<<<grid, 512, 0, stream>>>(weight, mask, A_bf, part);
    }
    // 3) reduce over g + bias
    {
        int n = Bsz * COLS;                                      // 360,448
        reduce_bias_kernel<<<(n + 255) / 256, 256, 0, stream>>>(part, bias, out);
    }
}
